// CaptioningRNN_32856499815011
// MI455X (gfx1250) — matrix operand order repaired and checked by reading, not compiled, not run on hardware
//
#include <hip/hip_runtime.h>
#include <hip/hip_bf16.h>
#include <math.h>

typedef __attribute__((ext_vector_type(16))) __bf16 v16bf;
typedef __attribute__((ext_vector_type(8)))  float  v8f;

#define WMMA_BF16(a, b, c) \
    __builtin_amdgcn_wmma_f32_16x16x32_bf16(false, (a), false, (b), (short)0, (c), false, false)

// A-matrix (bf16 16x32) per-lane K index for vector element j (half = lane>>4)
__device__ __forceinline__ int a_kidx(int j, int half) {
    return half * 16 + j;
}

// Inverse of a_kidx: for k5 = k%32, which (half, j) slot holds it.
// a_kidx bit4 == half, bits0-3 == j. Verified inverse for all 32 values.
__device__ __forceinline__ void a_pack_pos(int k5, int& half, int& j) {
    half = (k5 >> 4) & 1;
    j = k5 & 15;
}

// ---------------------------------------------------------------------------
// Kernel 0: pre-pack W_vocab (f32, [128][10000]) into bf16 in this file's (contiguous-k) WMMA
// B-operand register order: Bpack[ct][kc][lane][j] with
//   k = kc*32 + (lane>>4)*16 + j,  n = ct*16 + (lane&15)
// 625*4*32*16 bf16 = 2.56 MB (lives in L2 for all 512 WGs of kernel 2).
// ---------------------------------------------------------------------------
__global__ __launch_bounds__(256) void pack_wvocab_kernel(
    const float* __restrict__ W_vocab, __bf16* __restrict__ Bpack)
{
    const int idx = blockIdx.x * 256 + threadIdx.x;
    if (idx >= 625 * 4 * 32 * 16) return;
    const int j    = idx & 15;
    const int lane = (idx >> 4) & 31;
    const int kc   = (idx >> 9) & 3;
    const int ct   = idx >> 11;
    const int k = kc * 32 + (lane >> 4) * 16 + j;
    const int n = ct * 16 + (lane & 15);
    Bpack[idx] = (__bf16)W_vocab[(size_t)k * 10000 + n];
}

// ---------------------------------------------------------------------------
// Kernel 1: h0 = features@W_proj + b_proj, then 64 RNN steps
//   h_t = tanh(x_t @ W_ih^T + b_ih + h_{t-1} @ W_hh^T + b_hh)
// Grid: 8 WGs x 256 thr. WG b owns batch rows 16b..16b+15; wave w owns output
// columns 16w..16w+15. W_ih^T / W_hh^T slices persist in VGPRs for all steps.
// x_t and h tiles live in LDS in WMMA-A packed order [kc][lane][j] so each
// operand fetch is one aligned 32B vector LDS load.
// h_all is written in the same packed order, grouped by kernel-2 row tiles:
//   h_pack[row_tile][kc][lane2][j], row_tile = (n*64+t)/16, m = t&15.
// ---------------------------------------------------------------------------
__global__ __launch_bounds__(256) void rnn_fwd_kernel(
    const float* __restrict__ features, const int* __restrict__ captions,
    const float* __restrict__ W_proj, const float* __restrict__ b_proj,
    const float* __restrict__ embed, const float* __restrict__ W_ih,
    const float* __restrict__ W_hh, const float* __restrict__ b_ih,
    const float* __restrict__ b_hh, __bf16* __restrict__ h_pack)
{
    __shared__ __align__(32) __bf16 xs[4][32][16];  // x_t, WMMA-A packed
    __shared__ __align__(32) __bf16 hs[4][32][16];  // h_prev, WMMA-A packed

    const int n0   = blockIdx.x * 16;
    const int tid  = threadIdx.x;
    const int wave = tid >> 5;
    const int lane = tid & 31;
    const int half = lane >> 4;
    const int l15  = lane & 15;
    const int col  = wave * 16 + l15;   // output column (B/C N index)

    // Where this lane's output column lands inside a packed-A tile (k = col)
    int halfA, jA;
    a_pack_pos(col & 31, halfA, jA);
    const int kcA = col >> 5;

    // B = W^T slices: B[k][n] = W[n][k] -> contiguous per-lane global reads
    v16bf Bih[4], Bhh[4];
#pragma unroll
    for (int kc = 0; kc < 4; ++kc) {
        const int kb = kc * 32 + half * 16;
#pragma unroll
        for (int j = 0; j < 16; ++j) {
            Bih[kc][j] = (__bf16)W_ih[col * 128 + kb + j];
            Bhh[kc][j] = (__bf16)W_hh[col * 128 + kb + j];
        }
    }
    const float bias_sum = b_ih[col] + b_hh[col];
    const float bias0    = b_proj[col];

    // ---- h0 = features(16x512) @ W_proj(512x128) + b_proj ----
    v8f acc = {};
    for (int kc = 0; kc < 16; ++kc) {
        v16bf a, b;
#pragma unroll
        for (int j = 0; j < 16; ++j) {
            const int ka = kc * 32 + a_kidx(j, half);
            a[j] = (__bf16)features[(n0 + l15) * 512 + ka];
            const int kbb = kc * 32 + half * 16 + j;
            b[j] = (__bf16)W_proj[kbb * 128 + col];
        }
        acc = WMMA_BF16(a, b, acc);
    }
#pragma unroll
    for (int r = 0; r < 8; ++r) {
        const int brow = r + 8 * half;               // batch row (C/D layout)
        hs[kcA][brow + 16 * halfA][jA] = (__bf16)(acc[r] + bias0);
    }
    __syncthreads();

    // ---- 64 recurrent steps ----
    for (int t = 0; t < 64; ++t) {
        // gather x_t = embed[captions[:, t]] into packed-A LDS
        for (int e = tid; e < 16 * 128; e += 256) {
            const int i = e >> 7, k = e & 127;
            const int tok = captions[(n0 + i) * 65 + t];
            int hx, jx;
            a_pack_pos(k & 31, hx, jx);
            xs[k >> 5][i + 16 * hx][jx] = (__bf16)embed[tok * 128 + k];
        }
        __syncthreads();

        v8f hacc = {};
#pragma unroll
        for (int kc = 0; kc < 4; ++kc) {
            const v16bf ax = *(const v16bf*)&xs[kc][lane][0];
            const v16bf ah = *(const v16bf*)&hs[kc][lane][0];
            hacc = WMMA_BF16(ax, Bih[kc], hacc);
            hacc = WMMA_BF16(ah, Bhh[kc], hacc);
        }
        __syncthreads();   // all waves done reading hs

        const int m_t = t & 15;
        const size_t tile_base = ((size_t)(t >> 4) * 4 + kcA) * 32 * 16;  // + n*4 tiles
#pragma unroll
        for (int r = 0; r < 8; ++r) {
            const int brow = r + 8 * half;           // batch row
            const float v = tanhf(hacc[r] + bias_sum);
            const __bf16 vb = (__bf16)v;
            hs[kcA][brow + 16 * halfA][jA] = vb;
            const size_t rt = (size_t)(n0 + brow) * 4 * 4 * 32 * 16;  // n*4 row tiles
            h_pack[rt + tile_base + (size_t)(m_t + 16 * halfA) * 16 + jA] = vb;
        }
        __syncthreads();   // hs writes visible before next step
    }
}

// ---------------------------------------------------------------------------
// Kernel 2: scores = h@W_vocab + b_vocab, fused online log-softmax + NLL.
// Grid: 512 WGs x 256 thr; WG owns one 16-row tile (row_tile == blockIdx.x)
// of the 8192-row h matrix; waves stride the 625 column tiles. A and B are
// both pre-packed: one aligned 32B global load per lane per K-chunk.
// ---------------------------------------------------------------------------
__global__ __launch_bounds__(256) void scores_nll_kernel(
    const __bf16* __restrict__ h_pack, const int* __restrict__ captions,
    const __bf16* __restrict__ Bpack, const float* __restrict__ b_vocab,
    float* __restrict__ partials)
{
    __shared__ float smax[16][128];
    __shared__ float ssum[16][128];
    __shared__ float tscore[16];
    __shared__ int   ttarget[16];
    __shared__ float red[16];

    const int row0 = blockIdx.x * 16;   // global row = n*64 + t
    const int tid  = threadIdx.x;
    const int wave = tid >> 5;
    const int lane = tid & 31;
    const int half = lane >> 4;
    const int l15  = lane & 15;

    if (tid < 16) {
        const int row = row0 + tid;
        const int n = row >> 6, t = row & 63;
        ttarget[tid] = captions[n * 65 + t + 1];  // captions_out
        tscore[tid]  = 0.0f;
    }
    __syncthreads();

    // A = h row tile (16x128 bf16), already in WMMA-A packed order
    const v16bf* __restrict__ Hp = (const v16bf*)h_pack;  // index: (tile*4+kc)*32+lane
    v16bf A[4];
#pragma unroll
    for (int kc = 0; kc < 4; ++kc)
        A[kc] = Hp[((size_t)blockIdx.x * 4 + kc) * 32 + lane];

    int tgt[8];
#pragma unroll
    for (int r = 0; r < 8; ++r) tgt[r] = ttarget[r + 8 * half];

    float runM[8], runS[8];
#pragma unroll
    for (int r = 0; r < 8; ++r) { runM[r] = -__builtin_inff(); runS[r] = 0.0f; }

    const v16bf* __restrict__ Bp = (const v16bf*)Bpack;  // index: (ct*4+kc)*32+lane

    for (int ct = wave; ct < 625; ct += 8) {
        // prefetch next tile owned by this wave (global_prefetch_b8 path)
        const int nt = ct + 8;
        if (nt < 625) {
#pragma unroll
            for (int kc = 0; kc < 4; ++kc)
                __builtin_prefetch((const void*)&Bp[((size_t)nt * 4 + kc) * 32 + lane], 0, 3);
        }

        const int c0 = ct * 16;
        v8f acc = {};
#pragma unroll
        for (int kc = 0; kc < 4; ++kc) {
            const v16bf b = Bp[((size_t)ct * 4 + kc) * 32 + lane];
            acc = WMMA_BF16(A[kc], b, acc);
        }
        const float bias = b_vocab[c0 + l15];
#pragma unroll
        for (int r = 0; r < 8; ++r) {
            const float v = acc[r] + bias;
            if (c0 + l15 == tgt[r]) tscore[r + 8 * half] = v;  // exactly one lane matches
            if (v > runM[r]) {               // one exp per element
                runS[r] = runS[r] * __expf(runM[r] - v) + 1.0f;
                runM[r] = v;
            } else {
                runS[r] += __expf(v - runM[r]);
            }
        }
    }

#pragma unroll
    for (int r = 0; r < 8; ++r) {
        const int row = r + 8 * half;
        smax[row][wave * 16 + l15] = runM[r];
        ssum[row][wave * 16 + l15] = runS[r];
    }
    __syncthreads();

    if (tid < 16) {
        float M = -__builtin_inff(), S = 0.0f;
        for (int i = 0; i < 128; ++i) {
            const float m2 = smax[tid][i], s2 = ssum[tid][i];
            const float nm = fmaxf(M, m2);
            S = S * __expf(M - nm) + s2 * __expf(m2 - nm);
            M = nm;
        }
        const float nll  = -(tscore[tid] - M - __logf(S));
        const float mask = (ttarget[tid] != 0) ? 1.0f : 0.0f;
        red[tid] = nll * mask;
    }
    __syncthreads();
    if (tid == 0) {
        float s = 0.0f;
#pragma unroll
        for (int i = 0; i < 16; ++i) s += red[i];
        partials[blockIdx.x] = s;
    }
}

// ---------------------------------------------------------------------------
// Kernel 3: loss = sum(partials) / N
// ---------------------------------------------------------------------------
__global__ __launch_bounds__(256) void finalize_kernel(
    const float* __restrict__ partials, float* __restrict__ out)
{
    __shared__ float red[256];
    const int tid = threadIdx.x;
    float s = 0.0f;
    for (int i = tid; i < 512; i += 256) s += partials[i];
    red[tid] = s;
    __syncthreads();
    for (int off = 128; off > 0; off >>= 1) {
        if (tid < off) red[tid] += red[tid + off];
        __syncthreads();
    }
    if (tid == 0) out[0] = red[0] / 128.0f;
}

extern "C" void kernel_launch(void* const* d_in, const int* in_sizes, int n_in,
                              void* d_out, int out_size, void* d_ws, size_t ws_size,
                              hipStream_t stream) {
    (void)in_sizes; (void)n_in; (void)out_size; (void)ws_size;
    const float* features = (const float*)d_in[0];
    const int*   captions = (const int*)  d_in[1];
    const float* W_proj   = (const float*)d_in[2];
    const float* b_proj   = (const float*)d_in[3];
    const float* embed    = (const float*)d_in[4];
    const float* W_ih     = (const float*)d_in[5];
    const float* W_hh     = (const float*)d_in[6];
    const float* b_ih     = (const float*)d_in[7];
    const float* b_hh     = (const float*)d_in[8];
    const float* W_vocab  = (const float*)d_in[9];
    const float* b_vocab  = (const float*)d_in[10];

    char* ws = (char*)d_ws;
    __bf16* h_pack = (__bf16*)ws;                                  // 2 MB (packed)
    __bf16* Bpack  = (__bf16*)(ws + (size_t)128 * 64 * 128 * 2);   // 2.56 MB
    float*  partials = (float*)(ws + (size_t)128 * 64 * 128 * 2
                                   + (size_t)625 * 4 * 32 * 16 * 2);

    pack_wvocab_kernel<<<5000, 256, 0, stream>>>(W_vocab, Bpack);
    rnn_fwd_kernel<<<8, 256, 0, stream>>>(features, captions, W_proj, b_proj, embed,
                                          W_ih, W_hh, b_ih, b_hh, h_pack);
    scores_nll_kernel<<<512, 256, 0, stream>>>(h_pack, captions, Bpack, b_vocab, partials);
    finalize_kernel<<<1, 256, 0, stream>>>(partials, (float*)d_out);
}
